// IOProcessingAttention_7550552506819
// MI455X (gfx1250) — matrix operand order repaired and checked by reading, not compiled, not run on hardware
//
#include <hip/hip_runtime.h>

// Attention block: y = proj(softmax(mask(Q K^T / 8)) V)
// S^T = K Q^T and Y^T = Vt P^T so every WMMA fragment is two contiguous
// 16B-aligned b128 chunks. f16 in, f32 accumulate via v_wmma_f32_16x16x32_f16.
// K/V tiles staged with GLOBAL_LOAD_ASYNC_TO_LDS_B128 + double-buffered LDS.

#define TSEQ   2048
#define DMODEL 1024
#define NHEAD  16
#define HDIM   64
#define SKV    4096

typedef __attribute__((ext_vector_type(16))) _Float16 v16h;
typedef __attribute__((ext_vector_type(8)))  _Float16 h8;
typedef __attribute__((ext_vector_type(8)))  float    v8f;
typedef __attribute__((ext_vector_type(4)))  float    f4;

__device__ __forceinline__ v16h ld_frag(const _Float16* p0, const _Float16* p1) {
  h8 a = *(const h8*)p0;
  h8 b = *(const h8*)p1;
  return __builtin_shufflevector(a, b, 0,1,2,3,4,5,6,7,8,9,10,11,12,13,14,15);
}

// async 16B copy global -> LDS (per lane), tracked by ASYNCcnt
__device__ __forceinline__ void async_copy_b128(unsigned lds_off,
                                                unsigned long long gaddr) {
  asm volatile("global_load_async_to_lds_b128 %0, %1, off"
               :: "v"(lds_off), "v"(gaddr) : "memory");
}
__device__ __forceinline__ unsigned lds_off(const void* p) {
  return (unsigned)(uintptr_t)p;   // low 32 bits of flat LDS addr = DS address
}

// ---------------------------------------------------------------------------
// C = A[MxK] @ W[KxN] + bias.  A from f32 (A32) or f16 (A16).
// Output: f32 -> outf[.,1024]; else f16: col<1024 -> out0 row-major [.,1024],
// col>=1024 -> out1 TRANSPOSED [col][key] with row stride SKV (V^T layout).
__global__ void __launch_bounds__(128)
gemm64x64(const float* __restrict__ A32, const _Float16* __restrict__ A16,
          const float* __restrict__ W, const float* __restrict__ bias,
          _Float16* __restrict__ out0, _Float16* __restrict__ out1,
          float* __restrict__ outf, int M, int N, int K) {
  __shared__ __align__(16) _Float16 ldsA[64 * 40];   // [row][k] row-major
  __shared__ __align__(16) _Float16 ldsWt[64 * 40];  // [col][k] transposed
  const int tid  = threadIdx.x;
  const int lane = tid & 31;
  const int wave = tid >> 5;
  const int n    = lane & 15;
  const int hi   = lane >> 4;
  const int rowbase = blockIdx.y * 64;
  const int colbase = blockIdx.x * 64;

  v8f zero = {};
  v8f acc[4];
#pragma unroll
  for (int t = 0; t < 4; ++t) acc[t] = zero;

  const int ar = tid >> 1, ac = (tid & 1) * 16;   // A: row, 16-col chunk
  const int wc = tid >> 1, wk = (tid & 1) * 16;   // W: col, 16-k chunk

  h8 aLo = {}, aHi = {}, wLo = {}, wHi = {};
  auto load_stage = [&](int kk) {
    if (A32) {
      const float* src = A32 + (size_t)(rowbase + ar) * K + kk + ac;
      f4 x0 = *(const f4*)(src + 0), x1 = *(const f4*)(src + 4);
      f4 x2 = *(const f4*)(src + 8), x3 = *(const f4*)(src + 12);
#pragma unroll
      for (int i = 0; i < 4; ++i) { aLo[i] = (_Float16)x0[i]; aLo[i+4] = (_Float16)x1[i]; }
#pragma unroll
      for (int i = 0; i < 4; ++i) { aHi[i] = (_Float16)x2[i]; aHi[i+4] = (_Float16)x3[i]; }
    } else {
      const _Float16* src = A16 + (size_t)(rowbase + ar) * K + kk + ac;
      aLo = *(const h8*)(src); aHi = *(const h8*)(src + 8);
    }
#pragma unroll
    for (int i = 0; i < 8; ++i)
      wLo[i] = (_Float16)W[(size_t)(kk + wk + i) * N + colbase + wc];
#pragma unroll
    for (int i = 0; i < 8; ++i)
      wHi[i] = (_Float16)W[(size_t)(kk + wk + 8 + i) * N + colbase + wc];
  };

  load_stage(0);
  for (int kk = 0; kk < K; kk += 32) {
    *(h8*)&ldsA[ar * 40 + ac]       = aLo;
    *(h8*)&ldsA[ar * 40 + ac + 8]   = aHi;
    *(h8*)&ldsWt[wc * 40 + wk]      = wLo;
    *(h8*)&ldsWt[wc * 40 + wk + 8]  = wHi;
    __syncthreads();
    if (kk + 32 < K) load_stage(kk + 32);     // overlaps with compute below

    v16h a  = ld_frag(&ldsA[(wave * 16 + n) * 40 + 16 * hi],
                      &ldsA[(wave * 16 + n) * 40 + 16 * hi + 8]);
    v16h b0 = ld_frag(&ldsWt[(0 * 16 + n) * 40 + 16 * hi],
                      &ldsWt[(0 * 16 + n) * 40 + 16 * hi + 8]);
    v16h b1 = ld_frag(&ldsWt[(1 * 16 + n) * 40 + 16 * hi],
                      &ldsWt[(1 * 16 + n) * 40 + 16 * hi + 8]);
    v16h b2 = ld_frag(&ldsWt[(2 * 16 + n) * 40 + 16 * hi],
                      &ldsWt[(2 * 16 + n) * 40 + 16 * hi + 8]);
    v16h b3 = ld_frag(&ldsWt[(3 * 16 + n) * 40 + 16 * hi],
                      &ldsWt[(3 * 16 + n) * 40 + 16 * hi + 8]);
    acc[0] = __builtin_amdgcn_wmma_f32_16x16x32_f16(false, a, false, b0, (short)0, acc[0], false, false);
    acc[1] = __builtin_amdgcn_wmma_f32_16x16x32_f16(false, a, false, b1, (short)0, acc[1], false, false);
    acc[2] = __builtin_amdgcn_wmma_f32_16x16x32_f16(false, a, false, b2, (short)0, acc[2], false, false);
    acc[3] = __builtin_amdgcn_wmma_f32_16x16x32_f16(false, a, false, b3, (short)0, acc[3], false, false);
    __syncthreads();
  }

#pragma unroll
  for (int t = 0; t < 4; ++t) {
    const int col = colbase + t * 16 + n;
    const float bi = bias[col];
    if (outf) {
#pragma unroll
      for (int j = 0; j < 8; ++j) {
        int row = rowbase + wave * 16 + j + 8 * hi;
        outf[(size_t)row * DMODEL + col] = acc[t][j] + bi;
      }
    } else if (col < DMODEL) {
#pragma unroll
      for (int j = 0; j < 8; ++j) {
        int row = rowbase + wave * 16 + j + 8 * hi;
        out0[(size_t)row * DMODEL + col] = (_Float16)(acc[t][j] + bi);
      }
    } else {
      h8 v;
#pragma unroll
      for (int j = 0; j < 8; ++j) v[j] = (_Float16)(acc[t][j] + bi);
      int keybase = rowbase + wave * 16 + 8 * hi;
      *(h8*)(out1 + (size_t)(col - DMODEL) * SKV + keybase) = v;
    }
  }
}

// ---------------------------------------------------------------------------
// Flash attention, S^T formulation. Block = (64 queries, 1 head), 4 waves.
// K/V tiles (32 keys) double-buffered in LDS, filled by async global->LDS.
__global__ void __launch_bounds__(128)
attn_kernel(const _Float16* __restrict__ Q, const _Float16* __restrict__ Kb,
            const _Float16* __restrict__ Vt, _Float16* __restrict__ Y) {
  __shared__ __align__(16) _Float16 ldsK[2][32 * 40];   // [key][dim]
  __shared__ __align__(16) _Float16 ldsVt[2][64 * 40];  // [dim][key]
  const int tid  = threadIdx.x;
  const int lane = tid & 31;
  const int wave = tid >> 5;
  const int n    = lane & 15;
  const int hi   = lane >> 4;
  const int h    = blockIdx.y;
  const int qbase = blockIdx.x * 64;
  const int qg    = qbase + wave * 16 + n;   // this lane's query (B-frag col)

  const int kr = tid >> 2, kc = (tid & 3) * 16;   // K staging: key, dim chunk
  const int vd = tid >> 1, vc = (tid & 1) * 16;   // V staging: dim, key chunk

  // issue 4 async b128 copies for one 32-key tile into buffer `buf`
  auto issue_tile = [&](int kb, int buf) {
    unsigned long long ks = (unsigned long long)(uintptr_t)
        (Kb + (size_t)(kb + kr) * DMODEL + h * HDIM + kc);
    unsigned long long vs = (unsigned long long)(uintptr_t)
        (Vt + (size_t)(h * HDIM + vd) * SKV + kb + vc);
    async_copy_b128(lds_off(&ldsK[buf][kr * 40 + kc]),      ks);
    async_copy_b128(lds_off(&ldsK[buf][kr * 40 + kc + 8]),  ks + 16);
    async_copy_b128(lds_off(&ldsVt[buf][vd * 40 + vc]),     vs);
    async_copy_b128(lds_off(&ldsVt[buf][vd * 40 + vc + 8]), vs + 16);
  };

  issue_tile(0, 0);

  // B-frags of Q^T: element j<8 -> dim = kq*32 + j + 8*hi, j>=8 -> kq*32 + 16 + (j-8) + 8*hi (table)
  v16h bq[2];
#pragma unroll
  for (int kq = 0; kq < 2; ++kq) {
    const _Float16* p = Q + (size_t)qg * DMODEL + h * HDIM + kq * 32 + 8 * hi;
    bq[kq] = ld_frag(p, p + 16);
  }

  v8f zero = {};
  v8f o[4];                                  // Y^T frags: row=dim, col=query
#pragma unroll
  for (int t = 0; t < 4; ++t) o[t] = zero;
  float m = -1e30f, lsum = 0.0f;

  int cur = 0;
  const int kend = qbase + 64 + TSEQ;        // mask: key <= query + (T-1)
  for (int kb = 0; kb < kend; kb += 32) {
    if (kb + 32 < kend) {
      issue_tile(kb + 32, cur ^ 1);
      // 4 newly issued outstanding; older batch (this tile) completed
      asm volatile("s_wait_asynccnt 0x4" ::: "memory");
    } else {
      asm volatile("s_wait_asynccnt 0x0" ::: "memory");
    }
    __syncthreads();                         // tile `kb` visible to all waves
    const _Float16* lk = ldsK[cur];
    const _Float16* lv = ldsVt[cur];

    // S^T tiles: A = K rows (16 keys x 32 dims), B = Q^T
    v16h ak00 = ld_frag(&lk[(0 * 16 + n) * 40 + 8 * hi],      &lk[(0 * 16 + n) * 40 + 16 + 8 * hi]);
    v16h ak01 = ld_frag(&lk[(0 * 16 + n) * 40 + 32 + 8 * hi], &lk[(0 * 16 + n) * 40 + 48 + 8 * hi]);
    v16h ak10 = ld_frag(&lk[(1 * 16 + n) * 40 + 8 * hi],      &lk[(1 * 16 + n) * 40 + 16 + 8 * hi]);
    v16h ak11 = ld_frag(&lk[(1 * 16 + n) * 40 + 32 + 8 * hi], &lk[(1 * 16 + n) * 40 + 48 + 8 * hi]);
    v8f s[2]; s[0] = zero; s[1] = zero;
    s[0] = __builtin_amdgcn_wmma_f32_16x16x32_f16(false, ak00, false, bq[0], (short)0, s[0], false, false);
    s[0] = __builtin_amdgcn_wmma_f32_16x16x32_f16(false, ak01, false, bq[1], (short)0, s[0], false, false);
    s[1] = __builtin_amdgcn_wmma_f32_16x16x32_f16(false, ak10, false, bq[0], (short)0, s[1], false, false);
    s[1] = __builtin_amdgcn_wmma_f32_16x16x32_f16(false, ak11, false, bq[1], (short)0, s[1], false, false);

    // online softmax: key = kb + t*16 + j + 8*hi (row), query = n (lane)
    float p[2][8];
    float tmax = -1e30f;
#pragma unroll
    for (int t = 0; t < 2; ++t)
#pragma unroll
      for (int j = 0; j < 8; ++j) {
        int kg = kb + t * 16 + j + 8 * hi;
        float v = s[t][j] * 0.125f;
        v = (kg > qg + (TSEQ - 1)) ? -1e30f : v;
        p[t][j] = v;
        tmax = fmaxf(tmax, v);
      }
    tmax = fmaxf(tmax, __shfl_xor(tmax, 16, 32));
    float mnew  = fmaxf(m, tmax);
    float alpha = __expf(m - mnew);
    m = mnew;
    float psum = 0.0f;
#pragma unroll
    for (int t = 0; t < 2; ++t)
#pragma unroll
      for (int j = 0; j < 8; ++j) { p[t][j] = __expf(p[t][j] - mnew); psum += p[t][j]; }
    psum += __shfl_xor(psum, 16, 32);
    lsum = lsum * alpha + psum;
#pragma unroll
    for (int t = 0; t < 4; ++t)
#pragma unroll
      for (int j = 0; j < 8; ++j) o[t][j] *= alpha;

    // B-frag of P^T (32 keys x 16 queries), built in registers:
    // B[j]   (key=j+16*hi)   = hi ? xor16(p[1][j]) : p[0][j]
    // B[j+8] (key=j+8+16*hi) = hi ? p[1][j]        : xor16(p[0][j])
    v16h bp;
#pragma unroll
    for (int j = 0; j < 8; ++j) {
      float w0 = __shfl_xor(p[0][j], 16, 32);
      float w1 = __shfl_xor(p[1][j], 16, 32);
      bp[j]     = (_Float16)(hi ? w1 : p[0][j]);
      bp[j + 8] = (_Float16)(hi ? p[1][j] : w0);
    }

    // Y^T += Vt_tile @ P^T : A = Vt rows (16 dims x 32 keys)
    v16h av0 = ld_frag(&lv[(0 * 16 + n) * 40 + 16 * hi], &lv[(0 * 16 + n) * 40 + 16 * hi + 8]);
    v16h av1 = ld_frag(&lv[(1 * 16 + n) * 40 + 16 * hi], &lv[(1 * 16 + n) * 40 + 16 * hi + 8]);
    v16h av2 = ld_frag(&lv[(2 * 16 + n) * 40 + 16 * hi], &lv[(2 * 16 + n) * 40 + 16 * hi + 8]);
    v16h av3 = ld_frag(&lv[(3 * 16 + n) * 40 + 16 * hi], &lv[(3 * 16 + n) * 40 + 16 * hi + 8]);
    o[0] = __builtin_amdgcn_wmma_f32_16x16x32_f16(false, av0, false, bp, (short)0, o[0], false, false);
    o[1] = __builtin_amdgcn_wmma_f32_16x16x32_f16(false, av1, false, bp, (short)0, o[1], false, false);
    o[2] = __builtin_amdgcn_wmma_f32_16x16x32_f16(false, av2, false, bp, (short)0, o[2], false, false);
    o[3] = __builtin_amdgcn_wmma_f32_16x16x32_f16(false, av3, false, bp, (short)0, o[3], false, false);

    __syncthreads();                         // all done reading buf `cur`
    cur ^= 1;
  }

  // epilogue: o[t] row = dim t*16 + j + 8*hi, col = query n (this lane)
  const float inv = 1.0f / lsum;
#pragma unroll
  for (int t = 0; t < 4; ++t) {
    h8 v;
#pragma unroll
    for (int j = 0; j < 8; ++j) v[j] = (_Float16)(o[t][j] * inv);
    *(h8*)(Y + (size_t)qg * DMODEL + h * HDIM + t * 16 + 8 * hi) = v;
  }
}

// ---------------------------------------------------------------------------
extern "C" void kernel_launch(void* const* d_in, const int* in_sizes, int n_in,
                              void* d_out, int out_size, void* d_ws, size_t ws_size,
                              hipStream_t stream) {
  const float* l        = (const float*)d_in[0];
  const float* x        = (const float*)d_in[1];
  const float* W_attn_l = (const float*)d_in[2];
  const float* b_attn_l = (const float*)d_in[3];
  const float* W_attn_c = (const float*)d_in[4];
  const float* b_attn_c = (const float*)d_in[5];
  const float* W_q      = (const float*)d_in[6];
  const float* b_q      = (const float*)d_in[7];
  const float* W_proj   = (const float*)d_in[8];
  const float* b_proj   = (const float*)d_in[9];

  char* ws = (char*)d_ws;
  _Float16* Kbuf  = (_Float16*)(ws);                                   // 8 MB [key][1024]
  _Float16* Vtbuf = (_Float16*)(ws + (size_t)SKV * DMODEL * 2);        // 8 MB [1024][key]
  _Float16* Qbuf  = (_Float16*)(ws + (size_t)SKV * DMODEL * 4);        // 4 MB
  _Float16* Ybuf  = (_Float16*)(ws + (size_t)SKV * DMODEL * 4
                                   + (size_t)TSEQ * DMODEL * 2);       // 4 MB

  dim3 blk(128);
  gemm64x64<<<dim3(32, 32), blk, 0, stream>>>(l, nullptr, W_attn_l, b_attn_l,
      Kbuf, Vtbuf, nullptr, TSEQ, 2 * DMODEL, DMODEL);
  gemm64x64<<<dim3(32, 32), blk, 0, stream>>>(x, nullptr, W_attn_c, b_attn_c,
      Kbuf + (size_t)TSEQ * DMODEL, Vtbuf + TSEQ, nullptr,
      TSEQ, 2 * DMODEL, DMODEL);
  gemm64x64<<<dim3(16, 32), blk, 0, stream>>>(x, nullptr, W_q, b_q,
      Qbuf, nullptr, nullptr, TSEQ, DMODEL, DMODEL);
  attn_kernel<<<dim3(32, NHEAD), blk, 0, stream>>>(Qbuf, Kbuf, Vtbuf, Ybuf);
  gemm64x64<<<dim3(16, 32), blk, 0, stream>>>(nullptr, Ybuf, W_proj, b_proj,
      nullptr, nullptr, (float*)d_out, TSEQ, DMODEL, DMODEL);
}
